// FeaturePropagation_11802570130414
// MI455X (gfx1250) — matrix operand order repaired and checked by reading, not compiled, not run on hardware
//
#include <hip/hip_runtime.h>
#include <hip/hip_bf16.h>

// Problem dims (fixed by setup_inputs)
#define BATCH 2
#define NP1   4096     // coarse points
#define NP2   16384    // dense points
#define C1D   256      // features1 channels
#define C2D   128      // features2 channels
#define KX    384      // C1+C2 concat
#define HID   256      // hidden/out channels
#define BN_EPS 1e-3f

typedef __attribute__((ext_vector_type(16))) _Float16 v16h;  // 32 B
typedef __attribute__((ext_vector_type(8)))  _Float16 v8h;   // 16 B
typedef __attribute__((ext_vector_type(8)))  float    v8f;

// ---------------------------------------------------------------------------
// Prep: f32 -> f16 weight conversion
// ---------------------------------------------------------------------------
__global__ void cvt_f32_to_f16(const float* __restrict__ in,
                               _Float16* __restrict__ out, int n) {
  int i = blockIdx.x * 256 + threadIdx.x;
  if (i < n) out[i] = (_Float16)in[i];
}

// Prep: fold conv bias + inference BN into per-channel scale/shift
// y = (Wx + b - mean) * gamma/sqrt(var+eps) + beta  ==  (Wx)*s + t
__global__ void fold_bn_kernel(const float* __restrict__ bias,
                               const float* __restrict__ gamma,
                               const float* __restrict__ beta,
                               const float* __restrict__ mean,
                               const float* __restrict__ var,
                               float* __restrict__ s, float* __restrict__ t) {
  int i = threadIdx.x;  // H == 256 == blockDim.x
  float sc = gamma[i] * rsqrtf(var[i] + BN_EPS);
  s[i] = sc;
  t[i] = (bias[i] - mean[i]) * sc + beta[i];
}

// ---------------------------------------------------------------------------
// Phase 1: brute-force 3-NN (coarse cloud staged in 48 KB of LDS), inverse-
// distance^4 weighted interpolation, concat with features2, writing the
// activation tensor in WMMA-B packed layout:
//   Xp[b][blk=k/16][n][k%16]  (16 consecutive halves = 16 K-values of point n)
// ---------------------------------------------------------------------------
__global__ void three_nn_interp(const float* __restrict__ p1,   // [B,3,N1]
                                const float* __restrict__ p2,   // [B,3,N2]
                                const float* __restrict__ f1,   // [B,C1,N1]
                                const float* __restrict__ f2,   // [B,C2,N2]
                                _Float16* __restrict__ xp) {    // packed
  __shared__ float sx[NP1];
  __shared__ float sy[NP1];
  __shared__ float sz[NP1];

  const int b = blockIdx.x / (NP2 / 256);
  const int n = (blockIdx.x % (NP2 / 256)) * 256 + threadIdx.x;

  const float* pb = p1 + (size_t)b * 3 * NP1;
  for (int i = threadIdx.x; i < NP1; i += 256) {
    sx[i] = pb[i];
    sy[i] = pb[NP1 + i];
    sz[i] = pb[2 * NP1 + i];
  }
  __syncthreads();

  const float qx = p2[(size_t)b * 3 * NP2 + n];
  const float qy = p2[(size_t)b * 3 * NP2 + NP2 + n];
  const float qz = p2[(size_t)b * 3 * NP2 + 2 * NP2 + n];

  float d0 = 1e30f, d1 = 1e30f, d2v = 1e30f;
  int   i0 = 0,     i1 = 0,     i2 = 0;
#pragma unroll 4
  for (int j = 0; j < NP1; ++j) {
    const float dx = qx - sx[j];
    const float dy = qy - sy[j];
    const float dz = qz - sz[j];
    const float d = dx * dx + dy * dy + dz * dz;
    if (d < d0)       { d2v = d1; i2 = i1; d1 = d0; i1 = i0; d0 = d; i0 = j; }
    else if (d < d1)  { d2v = d1; i2 = i1; d1 = d;  i1 = j; }
    else if (d < d2v) { d2v = d;  i2 = j; }
  }

  // Reference: dist = (dist2)^2, clamp 1e-10, inverse, normalize.
  float a0 = fmaxf(d0, 0.f), a1 = fmaxf(d1, 0.f), a2 = fmaxf(d2v, 0.f);
  a0 = fmaxf(a0 * a0, 1e-10f);
  a1 = fmaxf(a1 * a1, 1e-10f);
  a2 = fmaxf(a2 * a2, 1e-10f);
  const float v0 = 1.f / a0, v1 = 1.f / a1, v2 = 1.f / a2;
  const float inv = 1.f / (v0 + v1 + v2);
  const float w0 = v0 * inv, w1 = v1 * inv, w2 = v2 * inv;

  const float* fb  = f1 + (size_t)b * C1D * NP1;
  const float* f2b = f2 + (size_t)b * C2D * NP2 + n;
  // chunk address for block blk: (((b*(KX/16) + blk) * NP2) + n) * 16
  _Float16* xo = xp + ((size_t)b * (KX / 16) * NP2 + n) * 16;

  // Interpolated features1 channels: blocks 0..15
  for (int blk = 0; blk < C1D / 16; ++blk) {
    v16h h;
#pragma unroll
    for (int ki = 0; ki < 16; ++ki) {
      const float* row = fb + (size_t)(blk * 16 + ki) * NP1;
      h[ki] = (_Float16)(w0 * row[i0] + w1 * row[i1] + w2 * row[i2]);
    }
    *(v16h*)(xo + (size_t)blk * NP2 * 16) = h;
  }
  // Copied features2 channels: blocks 16..23
  for (int blk = 0; blk < C2D / 16; ++blk) {
    v16h h;
#pragma unroll
    for (int ki = 0; ki < 16; ++ki)
      h[ki] = (_Float16)f2b[(size_t)(blk * 16 + ki) * NP2];
    *(v16h*)(xo + (size_t)(C1D / 16 + blk) * NP2 * 16) = h;
  }
}

// ---------------------------------------------------------------------------
// WMMA GEMM: y[b,m,n] = relu( (sum_k A[m,k]*X[b,k,n]) * s[m] + t[m] )
// One wave computes MT stacked 16x16 C tiles (64x16 for MT=4); the packed-
// layout B fragment is ONE contiguous 32-byte v16h load per lane, reused by
// MT WMMAs per K-step. A fragment: lane half grp holds k = grp*16 + j (same order as B).
// PACKED_OUT=true  -> store f16 in the same packed layout (feeds next layer)
// PACKED_OUT=false -> store f32 in standard [B][M][N] (final output)
// ---------------------------------------------------------------------------
template <int K, int MT, bool PACKED_OUT>
__global__ void wmma_gemm_bn_relu(const _Float16* __restrict__ Ah,  // [M][K]
                                  const _Float16* __restrict__ Xp,  // packed
                                  const float* __restrict__ s,
                                  const float* __restrict__ t,
                                  void* __restrict__ outv,
                                  int M, int N, int tiles_per_batch) {
  const int wave = threadIdx.x >> 5;
  const int lane = threadIdx.x & 31;
  const int tid  = blockIdx.x * 8 + wave;
  const int b = tid / tiles_per_batch;
  const int r = tid % tiles_per_batch;
  const int nt  = N >> 4;
  const int mb0 = (r / nt) * (16 * MT);
  const int nb  = (r % nt) << 4;

  const int row = lane & 15;   // M (for A) or N (for B/C) within tile
  const int grp = lane >> 4;

  const _Float16* Xb = Xp + (size_t)b * (K / 16) * N * 16;
  v8f acc[MT] = {};

#pragma unroll
  for (int kb = 0; kb < K; kb += 32) {
    // B fragment: lane -> N = nb+row, K = kb + grp*16 + j  (one 32B load)
    const v16h bf =
        *(const v16h*)(Xb + ((size_t)(kb / 16 + grp) * N + nb + row) * 16);
#pragma unroll
    for (int i = 0; i < MT; ++i) {
      v16h a;
      const _Float16* arow = Ah + (size_t)(mb0 + i * 16 + row) * K + kb;
#pragma unroll
      for (int j = 0; j < 16; ++j) {
        const int k = grp * 16 + j;
        a[j] = arow[k];
      }
      acc[i] = __builtin_amdgcn_wmma_f32_16x16x32_f16(
          /*neg_a=*/false, a, /*neg_b=*/false, bf,
          /*c_mod=*/(short)0, acc[i], /*reuse_a=*/false, /*reuse_b=*/false);
    }
  }

  // C tile layout: VGPR rr, lane -> M = mtile*16 + grp*8 + rr, N = nb + row
  if (PACKED_OUT) {
    _Float16* Yp = (_Float16*)outv;
#pragma unroll
    for (int i = 0; i < MT; ++i) {
      const int mtile = (mb0 >> 4) + i;
      v8h h;
#pragma unroll
      for (int rr = 0; rr < 8; ++rr) {
        const int m = mb0 + i * 16 + grp * 8 + rr;
        float v = acc[i][rr] * s[m] + t[m];
        h[rr] = (_Float16)fmaxf(v, 0.f);
      }
      // 8 consecutive halves of chunk (mtile, nb+row): one 16B store
      *(v8h*)(Yp + ((size_t)(b * (M / 16) + mtile) * N + nb + row) * 16 +
              grp * 8) = h;
    }
  } else {
    float* O = (float*)outv + (size_t)b * M * N + nb + row;
#pragma unroll
    for (int i = 0; i < MT; ++i) {
#pragma unroll
      for (int rr = 0; rr < 8; ++rr) {
        const int m = mb0 + i * 16 + grp * 8 + rr;
        float v = acc[i][rr] * s[m] + t[m];
        O[(size_t)m * N] = fmaxf(v, 0.f);
      }
    }
  }
}

// ---------------------------------------------------------------------------
extern "C" void kernel_launch(void* const* d_in, const int* in_sizes, int n_in,
                              void* d_out, int out_size, void* d_ws, size_t ws_size,
                              hipStream_t stream) {
  const float* p1  = (const float*)d_in[0];   // [2,3,4096]
  const float* p2  = (const float*)d_in[1];   // [2,3,16384]
  const float* f1  = (const float*)d_in[2];   // [2,256,4096]
  const float* f2  = (const float*)d_in[3];   // [2,128,16384]
  const float* W1  = (const float*)d_in[4];   // [256,384]
  const float* b1  = (const float*)d_in[5];
  const float* g1  = (const float*)d_in[6];
  const float* be1 = (const float*)d_in[7];
  const float* m1  = (const float*)d_in[8];
  const float* v1  = (const float*)d_in[9];
  const float* W2  = (const float*)d_in[10];  // [256,256]
  const float* b2  = (const float*)d_in[11];
  const float* g2  = (const float*)d_in[12];
  const float* be2 = (const float*)d_in[13];
  const float* m2  = (const float*)d_in[14];
  const float* v2  = (const float*)d_in[15];

  // Workspace carve-up (offsets are 32B-aligned)
  char* ws = (char*)d_ws;
  _Float16* W1h  = (_Float16*)ws;                         // 256*384 halves
  _Float16* W2h  = W1h + (size_t)HID * KX;                // 256*256 halves
  float*    s1   = (float*)(W2h + (size_t)HID * HID);
  float*    t1   = s1 + HID;
  float*    s2   = t1 + HID;
  float*    t2   = s2 + HID;
  _Float16* xcat = (_Float16*)(t2 + HID);                 // packed [2][24][16384][16]
  _Float16* y1h  = xcat + (size_t)BATCH * KX * NP2;       // packed [2][16][16384][16]
  (void)ws_size; (void)in_sizes; (void)n_in; (void)out_size;

  // Prep: weight conversion + BN folding
  cvt_f32_to_f16<<<(HID * KX + 255) / 256, 256, 0, stream>>>(W1, W1h, HID * KX);
  cvt_f32_to_f16<<<(HID * HID + 255) / 256, 256, 0, stream>>>(W2, W2h, HID * HID);
  fold_bn_kernel<<<1, HID, 0, stream>>>(b1, g1, be1, m1, v1, s1, t1);
  fold_bn_kernel<<<1, HID, 0, stream>>>(b2, g2, be2, m2, v2, s2, t2);

  // Phase 1: 3-NN + interpolation + concat -> packed f16 activations
  three_nn_interp<<<BATCH * (NP2 / 256), 256, 0, stream>>>(p1, p2, f1, f2, xcat);

  // Phase 2/3: WMMA GEMM + fused BN + ReLU  (MT=4 -> 64x16 C per wave)
  constexpr int MT = 4;
  const int tiles_per_batch = (HID / (16 * MT)) * (NP2 / 16);  // 4 * 1024
  const int total_tiles = BATCH * tiles_per_batch;             // 8192
  wmma_gemm_bn_relu<KX, MT, true><<<total_tiles / 8, 256, 0, stream>>>(
      W1h, xcat, s1, t1, (void*)y1h, HID, NP2, tiles_per_batch);
  wmma_gemm_bn_relu<HID, MT, false><<<total_tiles / 8, 256, 0, stream>>>(
      W2h, y1h, s2, t2, d_out, HID, NP2, tiles_per_batch);
}
